// ClassAwareKnittingMemoryV2_78116865180344
// MI455X (gfx1250) — matrix operand order repaired and checked by reading, not compiled, not run on hardware
//
#include <hip/hip_runtime.h>
#include <math.h>

// ---------------------------------------------------------------------------
// CDNA5 WMMA / TDM types
// ---------------------------------------------------------------------------
typedef __attribute__((ext_vector_type(16))) __bf16 v16bf;
typedef __attribute__((ext_vector_type(8)))  float  v8f;
typedef unsigned int u32x4 __attribute__((ext_vector_type(4)));
typedef int          i32x4 __attribute__((ext_vector_type(4)));
typedef int          i32x8 __attribute__((ext_vector_type(8)));

__device__ __forceinline__ unsigned short f2bf(float f) {
  unsigned int u = __builtin_bit_cast(unsigned int, f);
  u += 0x7FFFu + ((u >> 16) & 1u);   // round-to-nearest-even
  return (unsigned short)(u >> 16);
}
__device__ __forceinline__ unsigned pack2bf(float lo, float hi) {
  return (unsigned)f2bf(lo) | ((unsigned)f2bf(hi) << 16);
}

union Frag16 {
  v16bf v;
  unsigned int   w[8];
  unsigned short u[16];
};

__device__ __forceinline__ v8f wmma_bf16(v16bf a, v16bf b, v8f c) {
  return __builtin_amdgcn_wmma_f32_16x16x32_bf16(false, a, false, b, (short)0, c,
                                                 false, false);
}

// ---------------------------------------------------------------------------
// TDM: async-load a 2D bf16 tile (32 wide x 64 rows) into LDS.
// Descriptor packing per CDNA5 ISA.
// OOB rows/cols (beyond rem_k / rem_m) are zero-filled by the TDM.
// ---------------------------------------------------------------------------
__device__ __forceinline__ void tdm_load_a_tile(unsigned lds_off,
                                                const unsigned short* gptr,
                                                int rem_k, int rem_m,
                                                long lda_elems)
{
  unsigned long long ga = (unsigned long long)(uintptr_t)gptr;
  u32x4 g0;
  g0.x = 1u;                                        // count=1 (valid user D#)
  g0.y = lds_off;                                   // lds_addr (bytes)
  g0.z = (unsigned)(ga & 0xFFFFFFFFu);              // global_addr[31:0]
  g0.w = (unsigned)((ga >> 32) & 0x01FFFFFFu)       // global_addr[56:32]
       | 0x80000000u;                               // type=2 ("image")
  unsigned tk = (unsigned)rem_k;                    // tensor_dim0 (elements)
  unsigned tm = (unsigned)rem_m;                    // tensor_dim1
  unsigned long long st = (unsigned long long)lda_elems;
  i32x8 g1;
  g1[0] = 0x10000;                                  // data_size=1 -> 2 bytes
  g1[1] = (int)(tk << 16);                          // tensor_dim0[15:0]
  g1[2] = (int)((tk >> 16) | (tm << 16));           // td0[31:16] | td1[15:0]
  g1[3] = (int)((tm >> 16) | (32u << 16));          // td1[31:16] | tile_dim0=32
  g1[4] = 64;                                       // tile_dim1=64, tile_dim2=0
  g1[5] = (int)(unsigned)(st & 0xFFFFFFFFu);        // stride0[31:0]
  g1[6] = (int)(unsigned)((st >> 32) & 0xFFFFu);    // stride0[47:32], stride1 lo=0
  g1[7] = 0;
  i32x4 gz4 = {0, 0, 0, 0};
  i32x8 gz8 = {0, 0, 0, 0, 0, 0, 0, 0};
  __builtin_amdgcn_tensor_load_to_lds(g0, g1, gz4, gz4, gz8, 0);
}

// A fragment (16-bit 16x32 layout, ISA 7.12.2) from a bf16 LDS tile:
// two 16-byte LDS loads, no conversion.
__device__ __forceinline__ v16bf afrag_lds(const unsigned short* At, int row, int hL) {
  Frag16 a;
  const unsigned short* rp = At + row * 32;
  uint4 lo = *(const uint4*)(rp + 8 * hL);        // k = 8h .. 8h+7
  uint4 hi = *(const uint4*)(rp + 16 + 8 * hL);   // k = 16+8h .. 16+8h+7
  a.w[0] = lo.x; a.w[1] = lo.y; a.w[2] = lo.z; a.w[3] = lo.w;
  a.w[4] = hi.x; a.w[5] = hi.y; a.w[6] = hi.z; a.w[7] = hi.w;
  return a.v;
}

// B fragment from bf16 B^T (N x K row-major): two 16-byte global loads.
// element e <-> k = k0 + 8*hL + e (e<8), k0 + 16 + 8*hL + (e-8) (e>=8).
__device__ __forceinline__ v16bf bfrag_bt(const unsigned short* Bt, long ldb,
                                          int ncol, int N, int k0, int hL) {
  Frag16 b;
  if (ncol < N) {
    const unsigned short* bp = Bt + (size_t)ncol * ldb + k0 + 8 * hL;
    uint4 lo = *(const uint4*)(bp);
    uint4 hi = *(const uint4*)(bp + 16);
    b.w[0] = lo.x; b.w[1] = lo.y; b.w[2] = lo.z; b.w[3] = lo.w;
    b.w[4] = hi.x; b.w[5] = hi.y; b.w[6] = hi.z; b.w[7] = hi.w;
  } else {
#pragma unroll
    for (int e = 0; e < 8; ++e) b.w[e] = 0u;
  }
  return b.v;
}

// ---------------------------------------------------------------------------
// Batched GEMM: C[z] = gate * (A[z] (MxK,bf16) @ B[z] + bias), B given as
// bf16 B^T (NxK). f32 accumulate; outputs f32 C and/or bf16 Cb (same ldc).
// K multiple of 32. Block = 8 waves (2M x 4N), block tile 64x128, wave tile
// 32x32 -> 4 WMMAs/K-step. A tile staged by the Tensor Data Mover (wave 0),
// overlapped with B global loads.
// ---------------------------------------------------------------------------
__global__ __launch_bounds__(256) void gemm_bf16_wmma(
    const unsigned short* __restrict__ A, long lda, long sA,
    const unsigned short* __restrict__ Bt, long ldb, long sB,
    float* __restrict__ C, unsigned short* __restrict__ Cb, long ldc, long sC,
    int M, int N, int K,
    const float* __restrict__ bias, const float* __restrict__ gate)
{
  __shared__ unsigned short Atile[64 * 32];   // 4 KB bf16

  const int t     = threadIdx.x;
  const int lane  = t & 31;
  const int wave  = t >> 5;
  const int waveM = wave >> 2;                    // 0..1
  const int waveN = wave & 3;                     // 0..3
  const int z     = blockIdx.z;
  const int m0    = blockIdx.y * 64;
  const int n0w   = blockIdx.x * 128 + waveN * 32;

  A  += (size_t)z * sA;
  Bt += (size_t)z * sB;

  const int mL = lane & 15;
  const int hL = lane >> 4;
  const unsigned lds_off = (unsigned)(uintptr_t)(&Atile[0]);

  const int ncol0 = n0w + mL;
  const int ncol1 = n0w + 16 + mL;

  v8f acc00 = {}, acc01 = {}, acc10 = {}, acc11 = {};

  for (int k0 = 0; k0 < K; k0 += 32) {
    // async A-tile stage via Tensor Data Mover (one issue per block)
    if (wave == 0)
      tdm_load_a_tile(lds_off, A + (size_t)m0 * lda + k0, K - k0, M - m0, lda);

    // B fragments from global, overlapped with the TDM transfer
    v16bf b0 = bfrag_bt(Bt, ldb, ncol0, N, k0, hL);
    v16bf b1 = bfrag_bt(Bt, ldb, ncol1, N, k0, hL);

    if (wave == 0) __builtin_amdgcn_s_wait_tensorcnt(0);
    __syncthreads();

    v16bf a0 = afrag_lds(Atile, waveM * 32 + mL, hL);
    v16bf a1 = afrag_lds(Atile, waveM * 32 + 16 + mL, hL);

    acc00 = wmma_bf16(a0, b0, acc00);
    acc01 = wmma_bf16(a0, b1, acc01);
    acc10 = wmma_bf16(a1, b0, acc10);
    acc11 = wmma_bf16(a1, b1, acc11);

    __syncthreads();   // A tile consumed before next TDM overwrite
  }

  // ---- epilogue: optional bias (per column), sigmoid gate, dual outputs ----
  float g = 1.0f;
  if (gate) g = 1.0f / (1.0f + __expf(-gate[0]));

  auto store_tile = [&](v8f a, int rbase, int ncol) {
    if (ncol >= N) return;
    float bval = bias ? bias[ncol] : 0.0f;
#pragma unroll
    for (int r = 0; r < 8; ++r) {
      int row = rbase + r + 8 * hL;       // C layout: VGPR r -> M = r + 8*half
      if (row < M) {
        float val = (a[r] + bval) * g;
        if (C)  C [(size_t)z * sC + (size_t)row * ldc + ncol] = val;
        if (Cb) Cb[(size_t)z * sC + (size_t)row * ldc + ncol] = f2bf(val);
      }
    }
  };
  int rb = m0 + waveM * 32;
  store_tile(acc00, rb, ncol0);
  store_tile(acc01, rb, ncol1);
  store_tile(acc10, rb + 16, ncol0);
  store_tile(acc11, rb + 16, ncol1);
}

// ---------------------------------------------------------------------------
// Packed elementwise f32 -> bf16 (n must be even; i indexes pairs).
// ---------------------------------------------------------------------------
__global__ void convert_to_bf16(const float* __restrict__ src,
                                unsigned short* __restrict__ dst, size_t npairs)
{
  size_t i = (size_t)blockIdx.x * blockDim.x + threadIdx.x;
  if (i >= npairs) return;
  ((unsigned*)dst)[i] = pack2bf(src[2 * i], src[2 * i + 1]);
}

// ---------------------------------------------------------------------------
// Batched transpose-convert: dst[z][c, r] (bf16, R-major cols) = src[z][r, c].
// ---------------------------------------------------------------------------
__global__ void transpose_to_bf16(const float* __restrict__ src,
                                  unsigned short* __restrict__ dst,
                                  int R, int Cn, long sSrc, long sDst)
{
  int z = blockIdx.y;
  size_t i = (size_t)blockIdx.x * blockDim.x + threadIdx.x;
  size_t total = (size_t)R * Cn;
  if (i >= total) return;
  int r = (int)(i / Cn), c = (int)(i % Cn);
  dst[(size_t)z * sDst + (size_t)c * R + r] = f2bf(src[(size_t)z * sSrc + i]);
}

// ---------------------------------------------------------------------------
// Row softmax with temperature; writes bf16 attention weights.
// One wave per row; n <= 320.
// ---------------------------------------------------------------------------
__global__ __launch_bounds__(256) void softmax_rows(
    const float* __restrict__ data, unsigned short* __restrict__ out_bf,
    int rows, int n, const float* __restrict__ log_temp, float inv_sqrt_c)
{
  int row  = blockIdx.x * 8 + (threadIdx.x >> 5);
  int lane = threadIdx.x & 31;
  if (row >= rows) return;

  float temp  = __expf(log_temp[0]);
  temp        = fminf(fmaxf(temp, 0.05f), 1.0f);
  float scale = inv_sqrt_c / temp;

  const float* p = data + (size_t)row * n;
  unsigned short* o = out_bf + (size_t)row * n;
  float vals[10];
  int   cnt = 0;
  float mx  = -3.0e38f;
  for (int j = lane; j < n; j += 32) {
    float v = p[j] * scale;
    vals[cnt++] = v;
    mx = fmaxf(mx, v);
  }
  for (int of = 16; of > 0; of >>= 1) mx = fmaxf(mx, __shfl_xor(mx, of, 32));
  float sum = 0.f;
  for (int c = 0; c < cnt; ++c) { vals[c] = __expf(vals[c] - mx); sum += vals[c]; }
  for (int of = 16; of > 0; of >>= 1) sum += __shfl_xor(sum, of, 32);
  float inv = 1.0f / sum;
  cnt = 0;
  for (int j = lane; j < n; j += 32) o[j] = f2bf(vals[cnt++] * inv);
}

// ---------------------------------------------------------------------------
// kc_bf[b,v,s,:] (bf16) and vcf[b,v,s,:] (f32):
//   shared_{k,v}[v,s,:] + sum_r cat_A[cid,v,r] * catB{k,v}[v,r,s,:]
// Low-rank expansion replacing the 236-GFLOP cat_mem @ Wk/Wv GEMMs.
// ---------------------------------------------------------------------------
__global__ void build_catmem_kv(
    const float* __restrict__ sk, const float* __restrict__ sv,
    const float* __restrict__ cBk, const float* __restrict__ cBv,
    const float* __restrict__ cat_A, const int* __restrict__ cids,
    unsigned short* __restrict__ kc_bf, float* __restrict__ vc)
{
  int c     = blockIdx.x * 256 + threadIdx.x;   // 0..1023
  int rowid = blockIdx.y;                        // b*320 + v*32 + s
  int s  = rowid & 31;
  int bv = rowid >> 5;
  int v  = bv % 10;
  int b  = bv / 10;
  int cid = cids[b]; if (cid < 0) cid = 0;
  const float* arow = cat_A + ((size_t)cid * 10 + v) * 8;

  size_t sm_off = ((size_t)v * 32 + s) * 1024 + c;
  float ak = sk[sm_off];
  float av = sv[sm_off];
  size_t base = (((size_t)v * 8) * 32 + (size_t)s) * 1024 + c;
#pragma unroll
  for (int r = 0; r < 8; ++r) {
    float co = arow[r];
    ak += co * cBk[base + (size_t)r * 32 * 1024];
    av += co * cBv[base + (size_t)r * 32 * 1024];
  }
  size_t o = (size_t)rowid * 1024 + c;
  kc_bf[o] = f2bf(ak);
  vc[o]    = av;
}

// ---------------------------------------------------------------------------
// out[r,:] = src[gather?clip(gather[r]):r, :] @ W   (small: Nout = blockDim.x)
// ---------------------------------------------------------------------------
__global__ void proj_rows(const float* __restrict__ src, int ld_src,
                          const int* __restrict__ gather,
                          const float* __restrict__ W,
                          float* __restrict__ out, int K, int Nout)
{
  int r = blockIdx.x;
  int j = threadIdx.x;
  int sr = r;
  if (gather) { sr = gather[r]; if (sr < 0) sr = 0; }
  const float* s = src + (size_t)sr * ld_src;
  float acc = 0.f;
  for (int c = 0; c < K; ++c) acc += s[c] * W[(size_t)c * Nout + j];
  out[(size_t)r * Nout + j] = acc;
}

// ---------------------------------------------------------------------------
// alpha[b, s_begin+l] = sigmoid( gelu(qa + ce_part + scale_part + ab1) @ aW2
//                               + ab2 ) masked by category validity.
// ---------------------------------------------------------------------------
__global__ __launch_bounds__(256) void alpha_kernel(
    const float* __restrict__ qa, const float* __restrict__ ce_part,
    const float* __restrict__ sc_part_i, const float* __restrict__ ab1,
    const float* __restrict__ aW2, const float* __restrict__ ab2,
    const int* __restrict__ cids, float* __restrict__ alpha,
    int s_begin, int Ls)
{
  int row  = blockIdx.x * 8 + (threadIdx.x >> 5);
  int lane = threadIdx.x & 31;
  if (row >= 32 * Ls) return;
  int b = row / Ls, l = row % Ls;
  int memRow = b * 680 + s_begin + l;

  const float* qr = qa + (size_t)memRow * 128;
  const float* cp = ce_part + (size_t)b * 128;
  float acc = 0.f;
#pragma unroll
  for (int jj = 0; jj < 4; ++jj) {
    int j = lane + jj * 32;
    float xv = qr[j] + cp[j] + sc_part_i[j] + ab1[j];
    float hg = 0.5f * xv * (1.0f + erff(xv * 0.7071067811865475f)); // exact gelu
    acc += hg * aW2[j];
  }
  for (int o = 16; o > 0; o >>= 1) acc += __shfl_xor(acc, o, 32);
  if (lane == 0) {
    float a = 1.0f / (1.0f + __expf(-(acc + ab2[0])));
    if (cids[b] < 0) a = 0.0f;
    alpha[memRow] = a;
  }
}

// ---------------------------------------------------------------------------
// mem_bf = bf16( o_s + alpha * (o_c - o_s) ); os holds o_s (f32).
// ---------------------------------------------------------------------------
__global__ void combine_kernel(const float* __restrict__ os,
                               const float* __restrict__ oc,
                               const float* __restrict__ alpha,
                               unsigned short* __restrict__ mem_bf,
                               int s_begin, int Ls)
{
  size_t idx   = (size_t)blockIdx.x * blockDim.x + threadIdx.x;
  size_t total = (size_t)32 * Ls * 1024;
  if (idx >= total) return;
  int c = (int)(idx & 1023);
  int l = (int)((idx >> 10) % (size_t)Ls);
  int b = (int)(idx / ((size_t)Ls * 1024));
  int memRow = b * 680 + s_begin + l;
  float a  = alpha[memRow];
  float s  = os[(size_t)memRow * 1024 + c];
  float o  = oc[((size_t)(b * Ls + l)) * 1024 + c];
  mem_bf[(size_t)memRow * 1024 + c] = f2bf(s + a * (o - s));
}

__global__ void write_zeros(float* p, int n) {
  if ((int)threadIdx.x < n) p[threadIdx.x] = 0.f;
}

// ---------------------------------------------------------------------------
// Launch
// ---------------------------------------------------------------------------
extern "C" void kernel_launch(void* const* d_in, const int* in_sizes, int n_in,
                              void* d_out, int out_size, void* d_ws, size_t ws_size,
                              hipStream_t stream)
{
  (void)in_sizes; (void)n_in; (void)out_size; (void)ws_size;

  const float* x         = (const float*)d_in[0];
  const int*   cid       = (const int*)  d_in[1];
  const float* SM        = (const float*)d_in[2];
  const float* cat_A     = (const float*)d_in[3];
  const float* cat_B     = (const float*)d_in[4];
  const float* cat_emb   = (const float*)d_in[5];
  const float* scale_emb = (const float*)d_in[6];
  const float* Wq        = (const float*)d_in[7];
  const float* Wk        = (const float*)d_in[8];
  const float* Wv        = (const float*)d_in[9];
  const float* aW1       = (const float*)d_in[10];
  const float* ab1       = (const float*)d_in[11];
  const float* aW2       = (const float*)d_in[12];
  const float* ab2       = (const float*)d_in[13];
  const float* kW1       = (const float*)d_in[14];
  const float* kb1       = (const float*)d_in[15];
  const float* kW2       = (const float*)d_in[16];
  const float* kb2       = (const float*)d_in[17];
  const float* vW1       = (const float*)d_in[18];
  const float* vb1       = (const float*)d_in[19];
  const float* vW2       = (const float*)d_in[20];
  const float* vb2       = (const float*)d_in[21];
  const float* gk        = (const float*)d_in[22];
  const float* gv        = (const float*)d_in[23];
  const float* ltemp     = (const float*)d_in[24];

  const long BL = 21760;          // 32 * 680
  typedef unsigned short bh;

  char* base = (char*)d_ws;
  size_t off = 0;
  auto allocF = [&](size_t n) { float* p = (float*)(base + off);
                                off = (off + n * 4 + 63) & ~(size_t)63; return p; };
  auto allocH = [&](size_t n) { bh* p = (bh*)(base + off);
                                off = (off + n * 2 + 63) & ~(size_t)63; return p; };

  // f32 scratch
  float* sk    = allocF((size_t)320 * 1024);
  float* sv    = allocF((size_t)320 * 1024);
  float* cBk   = allocF((size_t)2560 * 1024);
  float* cBv   = allocF((size_t)2560 * 1024);
  float* vcf   = allocF((size_t)32 * 320 * 1024);
  float* qa    = allocF((size_t)BL * 128);
  float* cep   = allocF((size_t)32 * 128);
  float* scp   = allocF((size_t)10 * 128);
  float* logsS = allocF((size_t)32 * 256 * 320);
  float* logsC = allocF((size_t)32 * 256 * 320);
  float* ocb   = allocF((size_t)32 * 256 * 1024);
  float* osb   = allocF((size_t)BL * 1024);        // o_s at mem layout
  float* alpha = allocF((size_t)BL);
  // bf16 scratch
  bh* x_bf   = allocH((size_t)BL * 1024);
  bh* SM_bf  = allocH((size_t)320 * 1024);
  bh* cB_bf  = allocH((size_t)2560 * 1024);
  bh* q_bf   = allocH((size_t)BL * 1024);
  bh* sk_bf  = allocH((size_t)320 * 1024);
  bh* svT    = allocH((size_t)1024 * 320);
  bh* kc_bf  = allocH((size_t)32 * 320 * 1024);
  bh* vcT    = allocH((size_t)32 * 1024 * 320);
  bh* atS_bf = allocH((size_t)32 * 256 * 320);
  bh* atC_bf = allocH((size_t)32 * 256 * 320);
  bh* mem_bf = allocH((size_t)BL * 1024);
  bh* h1_bf  = allocH((size_t)BL * 64);
  bh* WqT    = allocH((size_t)1024 * 1024);
  bh* WkT    = allocH((size_t)1024 * 1024);
  bh* WvT    = allocH((size_t)1024 * 1024);
  bh* aW1T   = allocH((size_t)128 * 1024);
  bh* kW1T   = allocH((size_t)64 * 1024);
  bh* kW2T   = allocH((size_t)1024 * 64);
  bh* vW1T   = allocH((size_t)64 * 1024);
  bh* vW2T   = allocH((size_t)1024 * 64);

  auto gemm = [&](const bh* A, long lda, long sA,
                  const bh* Bt, long ldb, long sB,
                  float* C, bh* Cb, long ldc, long sC,
                  int M, int N, int K, int batches,
                  const float* bias, const float* gate) {
    dim3 grid((N + 127) / 128, (M + 63) / 64, batches);
    gemm_bf16_wmma<<<grid, 256, 0, stream>>>(A, lda, sA, Bt, ldb, sB,
                                             C, Cb, ldc, sC, M, N, K, bias, gate);
  };
  auto conv = [&](const float* s, bh* d, size_t n) {
    size_t np = n / 2;
    convert_to_bf16<<<(unsigned)((np + 255) / 256), 256, 0, stream>>>(s, d, np);
  };
  auto tconv = [&](const float* s, bh* d, int R, int Cn, long sS, long sD, int bz) {
    size_t tot = (size_t)R * Cn;
    dim3 grid((unsigned)((tot + 255) / 256), bz);
    transpose_to_bf16<<<grid, 256, 0, stream>>>(s, d, R, Cn, sS, sD);
  };

  // one-time operand conversions
  conv(x, x_bf, (size_t)BL * 1024);
  conv(SM, SM_bf, (size_t)320 * 1024);
  conv(cat_B, cB_bf, (size_t)2560 * 1024);
  tconv(Wq, WqT, 1024, 1024, 0, 0, 1);
  tconv(Wk, WkT, 1024, 1024, 0, 0, 1);
  tconv(Wv, WvT, 1024, 1024, 0, 0, 1);
  tconv(aW1, aW1T, 1024, 128, 0, 0, 1);        // first 1024 rows of aW1
  tconv(kW1, kW1T, 1024, 64, 0, 0, 1);
  tconv(kW2, kW2T, 64, 1024, 0, 0, 1);
  tconv(vW1, vW1T, 1024, 64, 0, 0, 1);
  tconv(vW2, vW2T, 64, 1024, 0, 0, 1);

  // query = x @ Wq  (bf16 out only)
  gemm(x_bf, 1024, 0, WqT, 1024, 0, nullptr, q_bf, 1024, 0, (int)BL, 1024, 1024, 1, nullptr, nullptr);
  // shared_k (f32 + bf16), shared_v (f32)
  gemm(SM_bf, 1024, 0, WkT, 1024, 0, sk, sk_bf, 1024, 0, 320, 1024, 1024, 1, nullptr, nullptr);
  gemm(SM_bf, 1024, 0, WvT, 1024, 0, sv, nullptr, 1024, 0, 320, 1024, 1024, 1, nullptr, nullptr);
  tconv(sv, svT, 320, 1024, 0, 0, 1);
  // low-rank precompute: cat_B @ Wk / Wv  (cat_B viewed as (2560,1024))
  gemm(cB_bf, 1024, 0, WkT, 1024, 0, cBk, nullptr, 1024, 0, 2560, 1024, 1024, 1, nullptr, nullptr);
  gemm(cB_bf, 1024, 0, WvT, 1024, 0, cBv, nullptr, 1024, 0, 2560, 1024, 1024, 1, nullptr, nullptr);
  // kc (bf16, key-major == B^T) / vc (f32) then transpose vc per batch
  build_catmem_kv<<<dim3(4, 10240), 256, 0, stream>>>(sk, sv, cBk, cBv, cat_A, cid, kc_bf, vcf);
  tconv(vcf, vcT, 320, 1024, 320L * 1024, 320L * 1024, 32);
  // alpha-MLP q-part:  qa = q @ aW1[0:1024,:]
  gemm(q_bf, 1024, 0, aW1T, 1024, 0, qa, nullptr, 128, 0, (int)BL, 128, 1024, 1, nullptr, nullptr);
  // alpha-MLP category / scale parts
  proj_rows<<<32, 128, 0, stream>>>(cat_emb, 1024, cid, aW1 + (size_t)1024 * 128, cep, 1024, 128);
  proj_rows<<<10, 128, 0, stream>>>(scale_emb, 1024, nullptr, aW1 + (size_t)2048 * 128, scp, 1024, 128);

  static const int PN[10] = {1, 2, 3, 4, 5, 6, 8, 10, 13, 16};
  int s_begin = 0;
  for (int i = 0; i < 10; ++i) {
    int Ls = PN[i] * PN[i];
    int V = i + 1, VS = V * 32;
    int rows = 32 * Ls;
    // shared attention: logits, softmax, o_s
    gemm(q_bf + (size_t)s_begin * 1024, 1024, 680L * 1024, sk_bf, 1024, 0,
         logsS, nullptr, VS, (long)Ls * VS, Ls, VS, 1024, 32, nullptr, nullptr);
    softmax_rows<<<(rows + 7) / 8, 256, 0, stream>>>(logsS, atS_bf, rows, VS, ltemp, 1.0f / 32.0f);
    gemm(atS_bf, VS, (long)Ls * VS, svT, 320, 0,
         osb + (size_t)s_begin * 1024, nullptr, 1024, 680L * 1024, Ls, 1024, VS, 32, nullptr, nullptr);
    // category attention (per-batch kc/vc)
    gemm(q_bf + (size_t)s_begin * 1024, 1024, 680L * 1024, kc_bf, 1024, 320L * 1024,
         logsC, nullptr, VS, (long)Ls * VS, Ls, VS, 1024, 32, nullptr, nullptr);
    softmax_rows<<<(rows + 7) / 8, 256, 0, stream>>>(logsC, atC_bf, rows, VS, ltemp, 1.0f / 32.0f);
    gemm(atC_bf, VS, (long)Ls * VS, vcT, 320, 320L * 1024,
         ocb, nullptr, 1024, (long)Ls * 1024, Ls, 1024, VS, 32, nullptr, nullptr);
    // alpha + blend -> mem_bf
    alpha_kernel<<<(rows + 7) / 8, 256, 0, stream>>>(qa, cep, scp + (size_t)i * 128,
                                                     ab1, aW2, ab2, cid, alpha, s_begin, Ls);
    size_t tot = (size_t)32 * Ls * 1024;
    combine_kernel<<<(unsigned)((tot + 255) / 256), 256, 0, stream>>>(osb, ocb, alpha, mem_bf, s_begin, Ls);
    s_begin += Ls;
  }

  float* outk = (float*)d_out;
  float* outv = outk + (size_t)BL * 1024;
  // mem_k = sigmoid(gk) * ((mem @ kW1 + kb1) @ kW2 + kb2)
  gemm(mem_bf, 1024, 0, kW1T, 1024, 0, nullptr, h1_bf, 64, 0, (int)BL, 64, 1024, 1, kb1, nullptr);
  gemm(h1_bf, 64, 0, kW2T, 64, 0, outk, nullptr, 1024, 0, (int)BL, 1024, 64, 1, kb2, gk);
  // mem_v = sigmoid(gv) * ((mem @ vW1 + vb1) @ vW2 + vb2)
  gemm(mem_bf, 1024, 0, vW1T, 1024, 0, nullptr, h1_bf, 64, 0, (int)BL, 64, 1024, 1, vb1, nullptr);
  gemm(h1_bf, 64, 0, vW2T, 64, 0, outv, nullptr, 1024, 0, (int)BL, 1024, 64, 1, vb2, gv);
  // trailing two scalar zeros of the output tuple
  write_zeros<<<1, 32, 0, stream>>>(outv + (size_t)BL * 1024, 2);
}
